// AnchorPoseModule_67104569032870
// MI455X (gfx1250) — compile-verified, matrix operand order checked by reading, not run on hardware
//
#include <hip/hip_runtime.h>
#include <hip/hip_bf16.h>

typedef __attribute__((ext_vector_type(16))) _Float16 v16h;
typedef __attribute__((ext_vector_type(8)))  float    v8f;

#define BLF   128      // BL = B*L frames
#define NJ    13
#define NPTS  20000
#define NS    500
#define BLJ   (BLF*NJ)            // 1664
#define MTOT  (BLJ*NS)            // 832000 pointnet rows
#define HID   64

static __device__ __forceinline__ float sigf(float x) { return 1.f / (1.f + __expf(-x)); }

// ---------------- stats helpers ----------------
__global__ void zero_kernel(float* p, int n) {
    int i = blockIdx.x * blockDim.x + threadIdx.x;
    if (i < n) p[i] = 0.f;
}
// blk layout: [sum(n), sq(n), mean(n), istd(n)]
__global__ void fin_kernel(float* blk, int nch, float cnt) {
    int t = threadIdx.x;
    if (t < nch) {
        float m = blk[t] / cnt;
        float v = blk[nch + t] / cnt - m * m;
        blk[2 * nch + t] = m;
        blk[3 * nch + t] = rsqrtf(v + 1e-5f);
    }
}

// ---------------- K0: radix-select top-NS nearest (smallest d2) ----------------
__launch_bounds__(256)
__global__ void topk_kernel(const float* __restrict__ x, const float* __restrict__ kpt,
                            int* __restrict__ idx) {
    __shared__ unsigned hist[256];
    __shared__ unsigned sh_prefix;
    __shared__ int sh_k, cless, ceq;
    int blj = blockIdx.x, tid = threadIdx.x;
    int bl = blj / NJ;
    float ax = kpt[blj * 2 + 0], ay = kpt[blj * 2 + 1];
    const float* base = x + (long)bl * NPTS * 28;
    if (tid == 0) { sh_prefix = 0u; sh_k = NS; cless = 0; ceq = 0; }
    __syncthreads();
    unsigned prefix = 0u; int krem = NS;
    for (int pass = 0; pass < 4; ++pass) {
        hist[tid] = 0u;
        __syncthreads();
        int shift = 24 - pass * 8;
        for (int n = tid; n < NPTS; n += 256) {
            float dx = base[(long)n * 28] - ax, dy = base[(long)n * 28 + 1] - ay;
            unsigned bu = __float_as_uint(dx * dx + dy * dy);
            unsigned u = (bu & 0x80000000u) ? ~bu : (bu | 0x80000000u);
            bool sel = (pass == 0) || ((u >> (shift + 8)) == (prefix >> (shift + 8)));
            if (sel) atomicAdd(&hist[(u >> shift) & 255], 1u);
        }
        __syncthreads();
        if (tid == 0) {
            unsigned cum = 0; int bsel = 255;
            for (int b = 0; b < 256; ++b) {
                if (cum + hist[b] >= (unsigned)sh_k) { bsel = b; break; }
                cum += hist[b];
            }
            sh_prefix |= ((unsigned)bsel << shift);
            sh_k -= (int)cum;
        }
        __syncthreads();
        prefix = sh_prefix; krem = sh_k;
        __syncthreads();
    }
    int nless = NS - krem;
    for (int n = tid; n < NPTS; n += 256) {
        float dx = base[(long)n * 28] - ax, dy = base[(long)n * 28 + 1] - ay;
        unsigned bu = __float_as_uint(dx * dx + dy * dy);
        unsigned u = (bu & 0x80000000u) ? ~bu : (bu | 0x80000000u);
        if (u < prefix) {
            int p = atomicAdd(&cless, 1);
            if (p < nless) idx[(long)blj * NS + p] = n;
        } else if (u == prefix) {
            int a = atomicAdd(&ceq, 1);
            if (a < krem) idx[(long)blj * NS + nless + a] = n;
        }
    }
}

// ---------------- K1: gather + build 32-ch f16 point features ----------------
__global__ void gather_kernel(const float* __restrict__ x, const float* __restrict__ kpt,
                              const int* __restrict__ idx, _Float16* __restrict__ gp) {
    long g = (long)blockIdx.x * blockDim.x + threadIdx.x;
    if (g >= (long)MTOT) return;
    int blj = (int)(g / NS), s = (int)(g % NS);
    int bl = blj / NJ;
    float ax = kpt[blj * 2 + 0], ay = kpt[blj * 2 + 1];
    int n = idx[(long)blj * NS + s];
    const float* pr = x + ((long)bl * NPTS + n) * 28;
    _Float16* o = gp + g * 32;
    o[0] = (_Float16)ax; o[1] = (_Float16)ay;
    o[2] = (_Float16)(pr[0] - ax); o[3] = (_Float16)(pr[1] - ay);
    for (int t = 0; t < 25; ++t) o[4 + t] = (_Float16)pr[3 + t];
    o[29] = (_Float16)pr[0]; o[30] = (_Float16)pr[1]; o[31] = (_Float16)pr[2];
}

// ---------------- PointNet GEMM: f16 WMMA, optional input BN+ReLU, stats out ----------------
template <int KIN, int NOUT, int KCH, bool BNIN>
__launch_bounds__(256)
__global__ void pn_gemm(const _Float16* __restrict__ in, const float* __restrict__ w,
                        const float* __restrict__ stin, const float* __restrict__ gg,
                        const float* __restrict__ bb, _Float16* __restrict__ out,
                        float* __restrict__ stout) {
    int wave = threadIdx.x >> 5, lane = threadIdx.x & 31;
    int lane16 = lane & 15, hi = lane >> 4;
    int m0 = (blockIdx.x * 8 + wave) * 16;
    if (m0 >= MTOT) return;
    long row = m0 + lane16;
    v16h A[KCH];
    for (int kc = 0; kc < KCH; ++kc)
        for (int e = 0; e < 16; ++e) {
            int k = kc * 32 + hi * 8 + e + (e < 8 ? 0 : 8);
            float v = 0.f;
            if (k < KIN) {
                v = (float)in[row * KIN + k];
                if (BNIN) {
                    v = (v - stin[2 * KIN + k]) * stin[3 * KIN + k] * gg[k] + bb[k];
                    v = v > 0.f ? v : 0.f;
                }
            }
            A[kc][e] = (_Float16)v;
        }
    for (int nt = 0; nt < NOUT / 16; ++nt) {
        int ncol = nt * 16 + lane16;
        v8f acc = {};
        for (int kc = 0; kc < KCH; ++kc) {
            v16h Bm;
            for (int e = 0; e < 16; ++e) {
                int k = kc * 32 + hi * 8 + e + (e < 8 ? 0 : 8);
                Bm[e] = (k < KIN) ? (_Float16)w[(long)ncol * KIN + k] : (_Float16)0.f;
            }
            acc = __builtin_amdgcn_wmma_f32_16x16x32_f16(false, A[kc], false, Bm,
                                                         (short)0, acc, false, false);
        }
        float s = 0.f, s2 = 0.f;
        for (int r = 0; r < 8; ++r) {
            float v = acc[r];
            s += v; s2 += v * v;
            out[(long)(m0 + r + hi * 8) * NOUT + ncol] = (_Float16)v;
        }
        atomicAdd(&stout[ncol], s);
        atomicAdd(&stout[NOUT + ncol], s2);
    }
}

// ---------------- K5: attention softmax pooling (bn3+relu applied on the fly) ----------------
__launch_bounds__(128)
__global__ void attn_kernel(const _Float16* __restrict__ act3, const float* __restrict__ st3,
                            const float* __restrict__ g3, const float* __restrict__ be3,
                            const float* __restrict__ wa, const float* __restrict__ ba,
                            float* __restrict__ attn_out, float* __restrict__ featT) {
    __shared__ float aw[NS];
    __shared__ float red[128];
    int blj = blockIdx.x, tid = threadIdx.x;
    const float* m  = st3 + 2 * 64;
    const float* is = st3 + 3 * 64;
    float lmax = -1e30f;
    for (int s = tid; s < NS; s += 128) {
        const _Float16* rowp = act3 + ((long)blj * NS + s) * 64;
        float acc = ba[0];
        for (int c = 0; c < 64; ++c) {
            float v = (float)rowp[c];
            v = (v - m[c]) * is[c] * g3[c] + be3[c];
            v = v > 0.f ? v : 0.f;
            acc += v * wa[c];
        }
        aw[s] = acc;
        lmax = fmaxf(lmax, acc);
    }
    red[tid] = lmax; __syncthreads();
    for (int o = 64; o; o >>= 1) { if (tid < o) red[tid] = fmaxf(red[tid], red[tid + o]); __syncthreads(); }
    float mx = red[0]; __syncthreads();
    float ls = 0.f;
    for (int s = tid; s < NS; s += 128) { float e = __expf(aw[s] - mx); aw[s] = e; ls += e; }
    red[tid] = ls; __syncthreads();
    for (int o = 64; o; o >>= 1) { if (tid < o) red[tid] += red[tid + o]; __syncthreads(); }
    float inv = 1.f / red[0]; __syncthreads();
    for (int s = tid; s < NS; s += 128) {
        float a = aw[s] * inv; aw[s] = a;
        attn_out[(long)blj * NS + s] = a;
    }
    __syncthreads();
    int c = tid & 63, half = tid >> 6;
    float acc = 0.f;
    for (int s = half; s < NS; s += 2) {
        float v = (float)act3[((long)blj * NS + s) * 64 + c];
        v = (v - m[c]) * is[c] * g3[c] + be3[c];
        v = v > 0.f ? v : 0.f;
        acc += aw[s] * v;
    }
    red[tid] = acc; __syncthreads();
    if (half == 0) {
        int b = blj / NJ, pos = blj % NJ;
        featT[((long)b * 64 + c) * NJ + pos] = red[tid] + red[tid + 64];
    }
}

// ---------------- PoseNet conv1d (VALID, ks=5), optional input BN+ReLU, stats out ----------------
template <int CIN, int COUT, int LIN, bool BNIN>
__global__ void conv_kernel(const float* __restrict__ in, const float* __restrict__ w,
                            const float* __restrict__ stin, const float* __restrict__ gin,
                            const float* __restrict__ bein, float* __restrict__ out,
                            float* __restrict__ stout) {
    const int LOUT = LIN - 4;
    int g = blockIdx.x * blockDim.x + threadIdx.x;
    if (g >= 128 * COUT * LOUT) return;
    int p = g % LOUT, o = (g / LOUT) % COUT, b = g / (LOUT * COUT);
    float acc = 0.f;
    for (int i = 0; i < CIN; ++i) {
        const float* ir = in + ((long)b * CIN + i) * LIN + p;
        const float* wr = w + ((long)o * CIN + i) * 5;
        for (int t = 0; t < 5; ++t) {
            float v = ir[t];
            if (BNIN) {
                v = (v - stin[2 * CIN + i]) * stin[3 * CIN + i] * gin[i] + bein[i];
                v = v > 0.f ? v : 0.f;
            }
            acc += v * wr[t];
        }
    }
    out[((long)b * COUT + o) * LOUT + p] = acc;
    atomicAdd(&stout[o], acc);
    atomicAdd(&stout[COUT + o], acc * acc);
}

__global__ void bnrelu_kernel(const float* __restrict__ in, const float* __restrict__ st,
                              const float* __restrict__ g, const float* __restrict__ be,
                              float* __restrict__ out, int n, int nch) {
    int i = blockIdx.x * blockDim.x + threadIdx.x;
    if (i >= n) return;
    int ch = i % nch;
    float v = (in[i] - st[2 * nch + ch]) * st[3 * nch + ch] * g[ch] + be[ch];
    out[i] = v > 0.f ? v : 0.f;
}

// ---------------- BiLSTM layer: grid.x = direction (0 fwd, 1 bwd) ----------------
__launch_bounds__(256)
__global__ void lstm_kernel(const float* __restrict__ xin, int din,
                            const float* wihF, const float* whhF, const float* bihF, const float* bhhF,
                            const float* wihB, const float* whhB, const float* bihB, const float* bhhB,
                            const float* __restrict__ h0, const float* __restrict__ c0, int layer,
                            float* __restrict__ yout, float* __restrict__ hn, float* __restrict__ cn) {
    __shared__ float h[512], c[512], z[2048], xsh[1024];
    int dir = blockIdx.x, tid = threadIdx.x;
    const float* wih = dir ? wihB : wihF;
    const float* whh = dir ? whhB : whhF;
    const float* bih = dir ? bihB : bihF;
    const float* bhh = dir ? bhhB : bhhF;
    int rowhc = 2 * layer + dir;
    for (int i = tid; i < 512; i += 256) { h[i] = h0[rowhc * 512 + i]; c[i] = c0[rowhc * 512 + i]; }
    __syncthreads();
    for (int stp = 0; stp < 16; ++stp) {
        int t = dir ? (15 - stp) : stp;
        for (int i = tid; i < 8 * din; i += 256)
            xsh[i] = xin[((long)(i / din) * 16 + t) * din + (i % din)];
        __syncthreads();
        for (int q = 0; q < 8; ++q) {
            int p = tid + 256 * q;
            int b = p >> 8, gi = p & 255;
            float acc = bih[gi] + bhh[gi];
            const float* xr = xsh + b * din;
            const float* wr = wih + (long)gi * din;
            for (int k = 0; k < din; ++k) acc += xr[k] * wr[k];
            const float* hr = h + b * 64;
            const float* wh = whh + gi * 64;
            for (int k = 0; k < 64; ++k) acc += hr[k] * wh[k];
            z[p] = acc;
        }
        __syncthreads();
        for (int q = 0; q < 2; ++q) {
            int p = tid + 256 * q;
            int b = p >> 6, j = p & 63;
            float zi = z[b * 256 + j], zf = z[b * 256 + 64 + j];
            float zg = z[b * 256 + 128 + j], zo = z[b * 256 + 192 + j];
            float cc = sigf(zf) * c[p] + sigf(zi) * tanhf(zg);
            float hh = sigf(zo) * tanhf(cc);
            c[p] = cc; h[p] = hh;
            yout[((long)b * 16 + t) * 128 + dir * 64 + j] = hh;
        }
        __syncthreads();
    }
    for (int i = tid; i < 512; i += 256) {
        hn[rowhc * 512 + i] = h[i];
        cn[rowhc * 512 + i] = c[i];
    }
}

// ---------------- FC head ----------------
__global__ void fc_kernel(const float* __restrict__ y, const float* __restrict__ w,
                          const float* __restrict__ b, float* __restrict__ out) {
    int g = blockIdx.x * blockDim.x + threadIdx.x;
    if (g >= 128 * 64) return;
    int o = g % 64, r = g / 64;
    float acc = b[o];
    for (int k = 0; k < 128; ++k) acc += y[(long)r * 128 + k] * w[(long)o * 128 + k];
    out[(long)r * 64 + o] = acc;
}

extern "C" void kernel_launch(void* const* d_in, const int* in_sizes, int n_in,
                              void* d_out, int out_size, void* d_ws, size_t ws_size,
                              hipStream_t stream) {
    (void)in_sizes; (void)n_in; (void)out_size; (void)ws_size;
    const float* x   = (const float*)d_in[0];
    const float* h0  = (const float*)d_in[3];
    const float* c0  = (const float*)d_in[4];
    const float* kpt = (const float*)d_in[5];
    // pointnet params
    const float* pn_w1 = (const float*)d_in[6];
    const float* pn_g1 = (const float*)d_in[8];
    const float* pn_be1 = (const float*)d_in[9];
    const float* pn_w2 = (const float*)d_in[10];
    const float* pn_g2 = (const float*)d_in[12];
    const float* pn_be2 = (const float*)d_in[13];
    const float* pn_w3 = (const float*)d_in[14];
    const float* pn_g3 = (const float*)d_in[16];
    const float* pn_be3 = (const float*)d_in[17];
    const float* pn_wa = (const float*)d_in[18];
    const float* pn_ba = (const float*)d_in[19];
    // posenet conv params
    const float* cv_w1 = (const float*)d_in[20];
    const float* cv_g1 = (const float*)d_in[22];
    const float* cv_be1 = (const float*)d_in[23];
    const float* cv_w2 = (const float*)d_in[24];
    const float* cv_g2 = (const float*)d_in[26];
    const float* cv_be2 = (const float*)d_in[27];
    const float* cv_w3 = (const float*)d_in[28];
    const float* cv_g3 = (const float*)d_in[30];
    const float* cv_be3 = (const float*)d_in[31];
    // lstm params: base 32, 8 tensors per layer (fwd wih,whh,bih,bhh; bwd same)
    const float* L[3][8];
    for (int l = 0; l < 3; ++l)
        for (int q = 0; q < 8; ++q) L[l][q] = (const float*)d_in[32 + l * 8 + q];
    const float* fc_w = (const float*)d_in[56];
    const float* fc_b = (const float*)d_in[57];

    // outputs: a_vec(8192) | attn_w(832000) | hn(3072) | cn(3072)
    float* out_avec = (float*)d_out;
    float* out_attn = out_avec + 8192;
    float* out_hn   = out_attn + (long)MTOT;
    float* out_cn   = out_hn + 3072;

    // workspace carve-out
    char* w8 = (char*)d_ws;
    size_t off = 0;
    auto carve = [&](size_t bytes) {
        void* p = w8 + off;
        off = (off + bytes + 255) & ~(size_t)255;
        return p;
    };
    int*       idxb = (int*)carve((size_t)MTOT * 4);
    _Float16*  R1   = (_Float16*)carve((size_t)MTOT * 48 * 2);   // gp(32ch) then act2(48ch)
    _Float16*  R2   = (_Float16*)carve((size_t)MTOT * 64 * 2);   // act1(32ch) then act3(64ch)
    float*     stats = (float*)carve(4096 * 4);
    float*     featT = (float*)carve((size_t)128 * 64 * NJ * 4);
    float*     cb1  = (float*)carve((size_t)128 * 96 * 9 * 4);
    float*     cb2  = (float*)carve((size_t)128 * 128 * 5 * 4);
    float*     cb3  = (float*)carve((size_t)128 * 64 * 4);
    float*     vbuf = (float*)carve((size_t)128 * 64 * 4);
    float*     yA   = (float*)carve((size_t)128 * 128 * 4);
    float*     yB   = (float*)carve((size_t)128 * 128 * 4);

    float* s1  = stats;         // 32-ch  [sum,sq,mean,istd]
    float* s2  = stats + 128;   // 48-ch
    float* s3  = stats + 320;   // 64-ch
    float* sc1 = stats + 576;   // 96-ch
    float* sc2 = stats + 960;   // 128-ch
    float* sc3 = stats + 1472;  // 64-ch

    zero_kernel<<<16, 256, 0, stream>>>(stats, 4096);

    topk_kernel<<<BLJ, 256, 0, stream>>>(x, kpt, idxb);
    gather_kernel<<<(MTOT + 255) / 256, 256, 0, stream>>>(x, kpt, idxb, R1);

    // PointNet: 3 x (WMMA GEMM + global BN)
    pn_gemm<32, 32, 1, false><<<MTOT / 16 / 8, 256, 0, stream>>>(R1, pn_w1, nullptr, nullptr, nullptr, R2, s1);
    fin_kernel<<<1, 256, 0, stream>>>(s1, 32, (float)MTOT);
    pn_gemm<32, 48, 1, true><<<MTOT / 16 / 8, 256, 0, stream>>>(R2, pn_w2, s1, pn_g1, pn_be1, R1, s2);
    fin_kernel<<<1, 256, 0, stream>>>(s2, 48, (float)MTOT);
    pn_gemm<48, 64, 2, true><<<MTOT / 16 / 8, 256, 0, stream>>>(R1, pn_w3, s2, pn_g2, pn_be2, R2, s3);
    fin_kernel<<<1, 256, 0, stream>>>(s3, 64, (float)MTOT);

    attn_kernel<<<BLJ, 128, 0, stream>>>(R2, s3, pn_g3, pn_be3, pn_wa, pn_ba, out_attn, featT);

    // PoseNet convs
    conv_kernel<64, 96, 13, false><<<(128 * 96 * 9 + 255) / 256, 256, 0, stream>>>(
        featT, cv_w1, nullptr, nullptr, nullptr, cb1, sc1);
    fin_kernel<<<1, 256, 0, stream>>>(sc1, 96, 128.f * 9.f);
    conv_kernel<96, 128, 9, true><<<(128 * 128 * 5 + 255) / 256, 256, 0, stream>>>(
        cb1, cv_w2, sc1, cv_g1, cv_be1, cb2, sc2);
    fin_kernel<<<1, 256, 0, stream>>>(sc2, 128, 128.f * 5.f);
    conv_kernel<128, 64, 5, true><<<(128 * 64 + 255) / 256, 256, 0, stream>>>(
        cb2, cv_w3, sc2, cv_g2, cv_be2, cb3, sc3);
    fin_kernel<<<1, 256, 0, stream>>>(sc3, 64, 128.f);
    bnrelu_kernel<<<(128 * 64 + 255) / 256, 256, 0, stream>>>(cb3, sc3, cv_g3, cv_be3, vbuf, 128 * 64, 64);

    // BiLSTM (3 layers, fwd+bwd blocks per launch)
    lstm_kernel<<<2, 256, 0, stream>>>(vbuf, 64,
        L[0][0], L[0][1], L[0][2], L[0][3], L[0][4], L[0][5], L[0][6], L[0][7],
        h0, c0, 0, yA, out_hn, out_cn);
    lstm_kernel<<<2, 256, 0, stream>>>(yA, 128,
        L[1][0], L[1][1], L[1][2], L[1][3], L[1][4], L[1][5], L[1][6], L[1][7],
        h0, c0, 1, yB, out_hn, out_cn);
    lstm_kernel<<<2, 256, 0, stream>>>(yB, 128,
        L[2][0], L[2][1], L[2][2], L[2][3], L[2][4], L[2][5], L[2][6], L[2][7],
        h0, c0, 2, yA, out_hn, out_cn);

    fc_kernel<<<(128 * 64 + 255) / 256, 256, 0, stream>>>(yA, fc_w, fc_b, out_avec);
}
